// QuantumAttentionLayer_4045859193502
// MI455X (gfx1250) — matrix operand order repaired and checked by reading, not compiled, not run on hardware
//
#include <hip/hip_runtime.h>
#include <hip/hip_bf16.h>

// ---------------------------------------------------------------------------
// QuantumAttentionLayer for MI455X (gfx1250, wave32, WMMA + TDM)
// B=4 S=256 D=2048 H=256 d=8
//   convert: x -> bf16, W -> transposed bf16 (one pass, lives in 192MB L2)
//   q/k/v/out GEMMs: 128x128 tiles, TDM (tensor_load_to_lds) double-buffered
//                    LDS staging (pad=4DW -> 80B rows, b128-aligned,
//                    bank-conflict-free), v_wmma_f32_16x16x32_bf16
//   attention: per-(b,s) 256x256 head-mixing scores (K=8) with
//              v_wmma_f32_16x16x4_f32, LDS softmax, attn@v f32 WMMA
// ---------------------------------------------------------------------------

typedef __attribute__((ext_vector_type(16))) __bf16   bf16x16;
typedef __attribute__((ext_vector_type(8)))  float    f32x8;
typedef __attribute__((ext_vector_type(2)))  float    f32x2;
typedef __attribute__((ext_vector_type(4)))  unsigned u32x4;
typedef __attribute__((ext_vector_type(8)))  unsigned u32x8;

#define TILE_ROW_U 20   // 16 DWORDs data + 4 DWORD TDM pad = 80B row stride

// ===========================================================================
// TDM: DMA a 2D bf16 tile (tile_d0 elems wide, tile_d1 rows, row stride in
// elements) from global memory into LDS, inserting 4 DWORDs of padding after
// every 16 DWORDs (-> LDS row stride = 20 DWORDs = 80B, 16B-aligned rows,
// conflict-free bank rotation). Issued by one wave, tracked with TENSORcnt.
// ===========================================================================
__device__ __forceinline__ void tdm_load_2d(unsigned lds_off,
                                            unsigned long long gaddr,
                                            unsigned tile_d0, unsigned tile_d1,
                                            unsigned long long row_stride_elems)
{
    u32x4 g0;
    g0[0] = 1u;                                            // count=1 (valid), load
    g0[1] = lds_off;                                       // lds_addr
    g0[2] = (unsigned)(gaddr & 0xFFFFFFFFu);               // global_addr[31:0]
    g0[3] = (unsigned)((gaddr >> 32) & 0x01FFFFFFu)        // global_addr[56:32]
          | (2u << 30);                                    // type=2 (image)

    u32x8 g1;
    // data_size=1 (2B) | pad_enable | pad_interval=3 (16 DW) | pad_amount=3 (4 DW)
    g1[0] = 0x06D10000u;
    g1[1] = (tile_d0 & 0xFFFFu) << 16;                     // tensor_dim0[15:0]
    g1[2] = (tile_d0 >> 16) | ((tile_d1 & 0xFFFFu) << 16); // tensor_dim0[31:16]|dim1[15:0]
    g1[3] = (tile_d1 >> 16) | ((tile_d0 & 0xFFFFu) << 16); // tensor_dim1[31:16]|tile_dim0
    g1[4] = (tile_d1 & 0xFFFFu);                           // tile_dim1 | tile_dim2=0
    g1[5] = (unsigned)(row_stride_elems & 0xFFFFFFFFu);    // tensor_dim0_stride[31:0]
    g1[6] = (unsigned)((row_stride_elems >> 32) & 0xFFFFu);// stride[47:32] | dim1_stride=0
    g1[7] = 0u;

    const u32x4 gz = {0u, 0u, 0u, 0u};                     // groups 2/3: unused dims

    asm volatile("tensor_load_to_lds %0, %1, %2, %3"
                 :
                 : "s"(g0), "s"(g1), "s"(gz), "s"(gz)
                 : "memory");
}

// ===========================================================================
// Converters (one pass; all converted operands then sit in L2 for the GEMMs)
// ===========================================================================
__global__ __launch_bounds__(256)
void cvt_f32_to_bf16(const float* __restrict__ in, __bf16* __restrict__ out, int n)
{
    const int i = (blockIdx.x * 256 + threadIdx.x) * 4;
    if (i < n) {
        const float4 f = *(const float4*)&in[i];
        out[i + 0] = (__bf16)f.x;
        out[i + 1] = (__bf16)f.y;
        out[i + 2] = (__bf16)f.z;
        out[i + 3] = (__bf16)f.w;
    }
}

// W[R][C] f32 (row-major, [in][out]) -> Wt[C][R] bf16 (row-major, [out][in])
__global__ __launch_bounds__(256)
void transpose_cvt_bf16(const float* __restrict__ W, __bf16* __restrict__ Wt,
                        int R, int Ccols)
{
    __shared__ float t[32][33];
    const int c0 = blockIdx.x * 32;
    const int r0 = blockIdx.y * 32;
    const int tx = threadIdx.x & 31;
    const int tw = threadIdx.x >> 5;          // 0..7
    #pragma unroll
    for (int i = 0; i < 4; ++i) {
        const int ty = tw * 4 + i;
        t[ty][tx] = W[(size_t)(r0 + ty) * Ccols + c0 + tx];
    }
    __syncthreads();
    #pragma unroll
    for (int i = 0; i < 4; ++i) {
        const int ty = tw * 4 + i;
        Wt[(size_t)(c0 + ty) * R + r0 + tx] = (__bf16)t[tx][ty];
    }
}

// ===========================================================================
// GEMM (TN): C[M,N] = A[M,K](bf16) @ Bt[N,K](bf16)^T + bias[N]  (f32 out)
// 128x128 block tile, 256 threads (8 waves), wave = 64x32 (4x2 wmma tiles).
// K chunks of 32, double-buffered TDM staging into 80B-stride LDS rows.
// ===========================================================================
__global__ __launch_bounds__(256)
void gemm_tn_bf16(const __bf16* __restrict__ A, const __bf16* __restrict__ Bt,
                  const float* __restrict__ bias, float* __restrict__ C,
                  int M, int N, int K)
{
    __shared__ unsigned AsU[2][128 * TILE_ROW_U];   // 2 x 10240 B
    __shared__ unsigned BsU[2][128 * TILE_ROW_U];   // 2 x 10240 B

    const int tid   = threadIdx.x;
    const int lane  = tid & 31;
    const int wave  = tid >> 5;
    const int waveM = wave >> 2;              // 0..1 -> 64 rows each
    const int waveN = wave & 3;               // 0..3 -> 32 cols each

    const int n0 = blockIdx.x * 128;
    const int m0 = blockIdx.y * 128;

    const int mn   = lane & 15;
    const int hsel = lane >> 4;               // 0/1 (lane half)

    const unsigned long long aBase =
        (unsigned long long)(uintptr_t)A + ((size_t)m0 * K) * 2ull;
    const unsigned long long bBase =
        (unsigned long long)(uintptr_t)Bt + ((size_t)n0 * K) * 2ull;
    const unsigned ldsA0 = (unsigned)(uintptr_t)&AsU[0][0];
    const unsigned ldsA1 = (unsigned)(uintptr_t)&AsU[1][0];
    const unsigned ldsB0 = (unsigned)(uintptr_t)&BsU[0][0];
    const unsigned ldsB1 = (unsigned)(uintptr_t)&BsU[1][0];

    f32x8 acc[4][2] = {};

    // ---- prologue: TDM chunk 0 into buffer 0 (wave 0 only) ----
    if (tid < 32) {
        tdm_load_2d(ldsA0, aBase, 32u, 128u, (unsigned long long)K);
        tdm_load_2d(ldsB0, bBase, 32u, 128u, (unsigned long long)K);
    }

    union Frag { bf16x16 v; u32x4 q[2]; };

    int buf = 0;
    for (int kk = 0; kk < K; kk += 32) {
        // ---- pipeline: issue next chunk, wait for current (wave 0) ----
        if (tid < 32) {
            if (kk + 32 < K) {
                const unsigned long long off = (size_t)(kk + 32) * 2ull;
                tdm_load_2d(buf ? ldsA0 : ldsA1, aBase + off, 32u, 128u,
                            (unsigned long long)K);
                tdm_load_2d(buf ? ldsB0 : ldsB1, bBase + off, 32u, 128u,
                            (unsigned long long)K);
                __builtin_amdgcn_s_wait_tensorcnt(2);   // current 2 done
            } else {
                __builtin_amdgcn_s_wait_tensorcnt(0);
            }
        }
        __syncthreads();

        // ---- fragment assembly: aligned b128 LDS loads from 80B rows ----
        const unsigned* As = AsU[buf];
        const unsigned* Bs = BsU[buf];
        Frag a[4], b[2];
        #pragma unroll
        for (int tm = 0; tm < 4; ++tm) {
            const unsigned* rp = As + (waveM * 64 + tm * 16 + mn) * TILE_ROW_U
                                + hsel * 4;      // A: K 0-7/16-23 vs 8-15/24-31
            a[tm].q[0] = *(const u32x4*)(rp);
            a[tm].q[1] = *(const u32x4*)(rp + 8);
        }
        #pragma unroll
        for (int tn = 0; tn < 2; ++tn) {
            const unsigned* rp = Bs + (waveN * 32 + tn * 16 + mn) * TILE_ROW_U
                                + hsel * 4;      // B: K 0-7/16-23 vs 8-15/24-31
            b[tn].q[0] = *(const u32x4*)(rp);
            b[tn].q[1] = *(const u32x4*)(rp + 8);
        }

        #pragma unroll
        for (int tm = 0; tm < 4; ++tm)
            #pragma unroll
            for (int tn = 0; tn < 2; ++tn)
                acc[tm][tn] = __builtin_amdgcn_wmma_f32_16x16x32_bf16(
                    false, a[tm].v, false, b[tn].v, (short)0, acc[tm][tn],
                    false, false);

        __syncthreads();     // protect buffer before it is refilled
        buf ^= 1;
    }

    // ---- epilogue: bias + store ----
    #pragma unroll
    for (int tm = 0; tm < 4; ++tm) {
        #pragma unroll
        for (int tn = 0; tn < 2; ++tn) {
            const int n  = n0 + waveN * 32 + tn * 16 + mn;
            const float bv = bias[n];
            #pragma unroll
            for (int r = 0; r < 8; ++r) {
                const int m = m0 + waveM * 64 + tm * 16 + r + hsel * 8;
                C[(size_t)m * N + n] = acc[tm][tn][r] + bv;
            }
        }
    }
}

// ===========================================================================
// Attention per (b,s): scores 256x256 over heads (K=8), softmax, attn@v.
// Grid: (B*S)*16 blocks of 256 threads; each block owns a 16-row strip.
// Emits bf16 context directly (feeds the final TDM GEMM with no conversion).
// ===========================================================================
__global__ __launch_bounds__(256)
void attn_kernel(const float* __restrict__ q, const float* __restrict__ k,
                 const float* __restrict__ v, const float* __restrict__ phase,
                 __bf16* __restrict__ ctx)
{
    __shared__ float Ssc[16][256];        // score strip (16 KB)
    __shared__ float Part[8][16][16];     // per-wave attn@v partials (8 KB)
    __shared__ float Red[16][16];
    __shared__ float RowStat[16];

    const int tid  = threadIdx.x;
    const int lane = tid & 31;
    const int wave = tid >> 5;

    const int bs = blockIdx.x >> 4;
    const int h0 = (blockIdx.x & 15) << 4;
    const int s  = bs & 255;              // phase broadcasts along the s dim

    const float phi  = phase[s];
    const float cphi = __cosf(phi), sphi = __sinf(phi);
    const float factor = (cphi * cphi + sphi * sphi) * 0.35355339059327373f;

    const float* qb = q + (size_t)bs * 2048;
    const float* kb = k + (size_t)bs * 2048;
    const float* vb = v + (size_t)bs * 2048;

    const int ml  = lane & 15;
    const int hs  = lane >> 4;
    const int kb0 = hs * 2;               // f32 WMMA A/B: lanes 16-31 hold K=2,3

    f32x2 aq0, aq1;
    {
        const float* qr = qb + (size_t)(h0 + ml) * 8;
        aq0 = *(const f32x2*)&qr[kb0];
        aq1 = *(const f32x2*)&qr[kb0 + 4];
    }

    // ---- scores: each wave computes 2 column tiles of 16 ----
    #pragma unroll
    for (int t = 0; t < 2; ++t) {
        const int g0 = (wave * 2 + t) << 4;
        const float* kr = kb + (size_t)(g0 + ml) * 8;
        const f32x2 b0 = *(const f32x2*)&kr[kb0];
        const f32x2 b1 = *(const f32x2*)&kr[kb0 + 4];
        f32x8 accS = {};
        accS = __builtin_amdgcn_wmma_f32_16x16x4_f32(
            false, aq0, false, b0, (short)0, accS, false, false);
        accS = __builtin_amdgcn_wmma_f32_16x16x4_f32(
            false, aq1, false, b1, (short)0, accS, false, false);
        #pragma unroll
        for (int r = 0; r < 8; ++r)
            Ssc[r + hs * 8][g0 + ml] = accS[r] * factor;
    }
    __syncthreads();

    // ---- row softmax (16 threads per row) ----
    const int row = tid >> 4;
    const int seg = tid & 15;
    {
        float mx = -3.4e38f;
        #pragma unroll
        for (int j = 0; j < 16; ++j) mx = fmaxf(mx, Ssc[row][seg * 16 + j]);
        Red[row][seg] = mx;
    }
    __syncthreads();
    if (seg == 0) {
        float mx = Red[row][0];
        #pragma unroll
        for (int j = 1; j < 16; ++j) mx = fmaxf(mx, Red[row][j]);
        RowStat[row] = mx;
    }
    __syncthreads();
    {
        const float mx = RowStat[row];
        float sum = 0.f;
        #pragma unroll
        for (int j = 0; j < 16; ++j) {
            const float e = __expf(Ssc[row][seg * 16 + j] - mx);
            Ssc[row][seg * 16 + j] = e;
            sum += e;
        }
        Red[row][seg] = sum;
    }
    __syncthreads();
    if (seg == 0) {
        float sm = 0.f;
        #pragma unroll
        for (int j = 0; j < 16; ++j) sm += Red[row][j];
        RowStat[row] = 1.0f / sm;
    }
    __syncthreads();
    {
        const float inv = RowStat[row];
        #pragma unroll
        for (int j = 0; j < 16; ++j) Ssc[row][seg * 16 + j] *= inv;
    }
    __syncthreads();

    // ---- ctx strip = attn(16x256) @ v(256x8, padded to 16 cols) ----
    {
        const float vmask = (ml < 8) ? 1.0f : 0.0f;   // arithmetic pad, no divergence
        const int   vcol  = ml & 7;
        f32x8 accC = {};
        const int kstart = wave * 32;
        #pragma unroll
        for (int kc = 0; kc < 32; kc += 4) {
            const int kg = kstart + kc + kb0;
            const f32x2 aa = *(const f32x2*)&Ssc[ml][kg];
            f32x2 bbv;
            bbv[0] = vb[(size_t)kg * 8 + vcol] * vmask;
            bbv[1] = vb[(size_t)(kg + 1) * 8 + vcol] * vmask;
            accC = __builtin_amdgcn_wmma_f32_16x16x4_f32(
                false, aa, false, bbv, (short)0, accC, false, false);
        }
        #pragma unroll
        for (int r = 0; r < 8; ++r)
            Part[wave][r + hs * 8][ml] = accC[r];
    }
    __syncthreads();

    {
        const int m = tid >> 4;
        const int n = tid & 15;
        if (n < 8) {
            float sum = 0.f;
            #pragma unroll
            for (int w = 0; w < 8; ++w) sum += Part[w][m][n];
            ctx[(size_t)bs * 2048 + (size_t)(h0 + m) * 8 + n] = (__bf16)sum;
        }
    }
}

// ===========================================================================
// Host-side launch
// ===========================================================================
extern "C" void kernel_launch(void* const* d_in, const int* in_sizes, int n_in,
                              void* d_out, int out_size, void* d_ws, size_t ws_size,
                              hipStream_t stream)
{
    const float* x  = (const float*)d_in[0];
    const float* ph = (const float*)d_in[1];
    const float* Wq = (const float*)d_in[2];
    const float* bq = (const float*)d_in[3];
    const float* Wk = (const float*)d_in[4];
    const float* bk = (const float*)d_in[5];
    const float* Wv = (const float*)d_in[6];
    const float* bv = (const float*)d_in[7];
    const float* Wo = (const float*)d_in[8];
    const float* bo = (const float*)d_in[9];
    float* out = (float*)d_out;

    const int M = 4 * 256;      // B*S = 1024
    const int N = 2048;
    const int K = 2048;

    char* w = (char*)d_ws;
    __bf16* xb  = (__bf16*)w;  w += (size_t)M * K * 2;          // 4 MB
    __bf16* Wqt = (__bf16*)w;  w += (size_t)K * N * 2;          // 8 MB
    __bf16* Wkt = (__bf16*)w;  w += (size_t)K * N * 2;
    __bf16* Wvt = (__bf16*)w;  w += (size_t)K * N * 2;
    __bf16* Wot = (__bf16*)w;  w += (size_t)K * N * 2;
    float*  qb  = (float*)w;   w += (size_t)M * N * 4;          // 8 MB
    float*  kbf = (float*)w;   w += (size_t)M * N * 4;
    float*  vbf = (float*)w;   w += (size_t)M * N * 4;
    __bf16* cb  = (__bf16*)w;  w += (size_t)M * N * 2;          // 4 MB

    // ---- one-pass precision/layout conversion ----
    cvt_f32_to_bf16<<<(M * K) / (256 * 4), 256, 0, stream>>>(x, xb, M * K);
    dim3 tg(N / 32, K / 32);
    transpose_cvt_bf16<<<tg, 256, 0, stream>>>(Wq, Wqt, K, N);
    transpose_cvt_bf16<<<tg, 256, 0, stream>>>(Wk, Wkt, K, N);
    transpose_cvt_bf16<<<tg, 256, 0, stream>>>(Wv, Wvt, K, N);
    transpose_cvt_bf16<<<tg, 256, 0, stream>>>(Wo, Wot, K, N);

    // ---- projections (TDM + bf16 WMMA) ----
    dim3 gg(N / 128, M / 128);  // (16, 8)
    gemm_tn_bf16<<<gg, 256, 0, stream>>>(xb, Wqt, bq, qb, M, N, K);
    gemm_tn_bf16<<<gg, 256, 0, stream>>>(xb, Wkt, bk, kbf, M, N, K);
    gemm_tn_bf16<<<gg, 256, 0, stream>>>(xb, Wvt, bv, vbf, M, N, K);

    // ---- head-mixing attention (f32 WMMA, K=8) ----
    attn_kernel<<<M * 16, 256, 0, stream>>>(qb, kbf, vbf, ph, cb);

    // ---- output projection ----
    gemm_tn_bf16<<<gg, 256, 0, stream>>>(cb, Wot, bo, out, M, N, K);
}
